// MPCA_34239479284066
// MI455X (gfx1250) — matrix operand order repaired and checked by reading, not compiled, not run on hardware
//
#include <hip/hip_runtime.h>
#include <hip/hip_bf16.h>
#include <math.h>

typedef __attribute__((ext_vector_type(16))) _Float16 v16h;
typedef __attribute__((ext_vector_type(8)))  _Float16 v8h;
typedef __attribute__((ext_vector_type(8)))  float    v8f;

#define C_IN   48
#define C3     144
#define HW     56
#define NPIX   (HW*HW)        // 3136
#define WW     49             // 7*7 window elements
#define TOK    9216           // 64 groups * 144 channels
#define DPAD   64             // feature dim padded 49 -> 64 for WMMA K

static __device__ __forceinline__ v8f wmma_f16(v16h a, v16h b, v8f c) {
  // D(f32 16x16) = A(f16 16x32) * B(f16 32x16) + C
  return __builtin_amdgcn_wmma_f32_16x16x32_f16(false, a, false, b, (short)0, c,
                                                false, false);
}

// -----------------------------------------------------------------------------
// Kernel 1: depthwise conv (3/5/7, SAME zero pad) fused with the reference's
// *linear* 49-flatten; output f16 [TOK][DPAD], pad features zeroed.
// token g = n*144 + oc ; feature k -> image pixel p = n*49 + k.
// -----------------------------------------------------------------------------
__global__ __launch_bounds__(64) void dwconv_pack(
    const float* __restrict__ x, const float* __restrict__ w3,
    const float* __restrict__ w5, const float* __restrict__ w7,
    _Float16* __restrict__ yf) {
  int g  = blockIdx.x;
  int n  = g / C3;
  int oc = g % C3;
  int br = oc / C_IN, c = oc % C_IN;
  int k  = threadIdx.x;
  _Float16* dst = yf + (size_t)g * DPAD;
  if (k >= WW) { dst[k] = (_Float16)0.0f; return; }
  int p = n * WW + k;
  int h = p / HW, w = p % HW;
  int ks = 3 + 2 * br, r = ks >> 1;
  const float* wk = (br == 0 ? w3 : (br == 1 ? w5 : w7)) + c * ks * ks;
  const float* xc = x + (size_t)c * NPIX;
  float acc = 0.f;
  for (int dy = 0; dy < ks; ++dy) {
    int hh = h + dy - r;
    if ((unsigned)hh >= HW) continue;
    for (int dx = 0; dx < ks; ++dx) {
      int ww2 = w + dx - r;
      if ((unsigned)ww2 >= HW) continue;
      acc += xc[hh * HW + ww2] * wk[dy * ks + dx];
    }
  }
  dst[k] = (_Float16)acc;
}

// -----------------------------------------------------------------------------
// Kernel 2: flash attention, no max-shift (|S/49| = O(1) here, exp() safe; the
// softmax is shift-invariant so the result is identical). Row sums of P are
// accumulated with a WMMA against an all-ones B matrix: the f32 C/D layout
// (lane L, reg r -> row r+8*(L/16), value replicated over the 16 columns)
// delivers per-row sums already replicated per lane group -> zero shuffles.
//
// 8 waves/WG share a double-buffered 32-key K/V chunk (row-major kt for the
// Q@K^T B-fragments, transposed ktT for the P@V B-fragments -> all operand
// reads are contiguous ds_load_b128). Each wave owns 32 queries (2 q-tiles).
// Keys are split into `nseg` segments (grid.y); without max-shift the partial
// (O~, l~) combine by plain addition (deterministic combine kernel).
//
// WMMA register layouts (wave32, CDNA5 ISA 7.12.2):
//   A (16x32 f16): lane L row M=L%16; elem i -> K = 16*(L/16)+i (this code's order)
//   B (32x16 f16): lane L col N=L%16; elem i -> K = 16*(L/16)+i (this code's order, same as A)
//   C/D (f32 16x16): lane L, reg r -> (M = r+8*(L/16), N = L%16)
// -----------------------------------------------------------------------------
__global__ __launch_bounds__(256) void flash_attn(
    const _Float16* __restrict__ yf, float* __restrict__ opart,
    float* __restrict__ lpart, int keys_per_seg) {
  __shared__ __align__(32) _Float16 kt [2][32 * DPAD];  // [key][feat]
  __shared__ __align__(32) _Float16 ktT[2][DPAD * 32];  // [feat][key]
  __shared__ __align__(32) _Float16 pt [8][16 * 32];    // per-wave P staging

  const int tid  = threadIdx.x;
  const int wave = tid >> 5, lane = tid & 31;
  const int lo = lane & 15, hi = lane >> 4;
  const int qbase = blockIdx.x * 256 + wave * 32;
  const int seg   = blockIdx.y;
  const int k0    = seg * keys_per_seg;
  const int nchunk = keys_per_seg >> 5;

  // Q fragments (A-layout) for the wave's two 16-query tiles
  v16h qa[4];
  #pragma unroll
  for (int qt = 0; qt < 2; ++qt) {
    const _Float16* qrow = yf + (size_t)(qbase + 16 * qt + lo) * DPAD;
    #pragma unroll
    for (int i = 0; i < 16; ++i) {
      int k = 16 * hi + i;
      qa[2 * qt + 0][i] = qrow[k];
      qa[2 * qt + 1][i] = qrow[32 + k];
    }
  }

  v8f o[8] = {};        // [qtile][ntile] f32 accumulators (unnormalized)
  v8f lsum[2] = {};     // row sums of P, via ones-WMMA
  v16h ones;
  #pragma unroll
  for (int i = 0; i < 16; ++i) ones[i] = (_Float16)1.0f;

  const float invww = 1.0f / 49.0f;
  const int srow = tid >> 3, squad = tid & 7;  // cooperative staging coords

  // prologue: stage chunk 0 into buffer 0 (row-major + transposed)
  {
    uint4 stg = *(const uint4*)(yf + (size_t)(k0 + srow) * DPAD + squad * 8);
    *(uint4*)(&kt[0][srow * DPAD + squad * 8]) = stg;
    const _Float16* s8 = (const _Float16*)&stg;
    #pragma unroll
    for (int j = 0; j < 8; ++j)
      ktT[0][(squad * 8 + j) * 32 + srow] = s8[j];
  }
  __syncthreads();

  for (int c = 0; c < nchunk; ++c) {
    const int buf = c & 1;
    // prefetch next chunk (global load overlaps compute below)
    uint4 nxt;
    const bool havenext = (c + 1 < nchunk);
    if (havenext)
      nxt = *(const uint4*)(yf + (size_t)(k0 + (c + 1) * 32 + srow) * DPAD + squad * 8);

    // B fragments for S = Q @ K^T : kb[t*2+ks], contiguous 32B LDS reads
    v16h kb[4];
    #pragma unroll
    for (int t = 0; t < 2; ++t)
      #pragma unroll
      for (int ks = 0; ks < 2; ++ks)
        kb[t * 2 + ks] =
            *(const v16h*)(&kt[buf][(16 * t + lo) * DPAD + ks * 32 + 16 * hi]);

    // per q-tile: S, exp, P staging, l-sum WMMA
    v16h pa[2];
    #pragma unroll
    for (int qt = 0; qt < 2; ++qt) {
      v8f s0 = {}, s1 = {};
      s0 = wmma_f16(qa[2 * qt + 0], kb[0], s0);
      s0 = wmma_f16(qa[2 * qt + 1], kb[1], s0);
      s1 = wmma_f16(qa[2 * qt + 0], kb[2], s1);
      s1 = wmma_f16(qa[2 * qt + 1], kb[3], s1);

      // P = exp(S/49) (no shift needed; see header comment)
      _Float16* ptw = &pt[wave][0];
      #pragma unroll
      for (int r = 0; r < 8; ++r) {
        int row = r + 8 * hi;
        ptw[row * 32 + lo]      = (_Float16)__expf(s0[r] * invww);
        ptw[row * 32 + 16 + lo] = (_Float16)__expf(s1[r] * invww);
      }
      __asm__ volatile("" ::: "memory");  // keep LDS store->load order
      // reload P in A-layout: two contiguous 16B runs per lane
      {
        const _Float16* prow = ptw + lo * 32;
        v8h x0 = *(const v8h*)(prow + 16 * hi);
        v8h x1 = *(const v8h*)(prow + 16 * hi + 8);
        pa[qt] = __builtin_shufflevector(x0, x1, 0, 1, 2, 3, 4, 5, 6, 7,
                                         8, 9, 10, 11, 12, 13, 14, 15);
      }
      lsum[qt] = wmma_f16(pa[qt], ones, lsum[qt]);
    }

    // O += P @ V : V B-fragments contiguous from transposed tile
    #pragma unroll
    for (int nt = 0; nt < 4; ++nt) {
      v16h vb = *(const v16h*)(&ktT[buf][(16 * nt + lo) * 32 + 16 * hi]);
      o[0 * 4 + nt] = wmma_f16(pa[0], vb, o[0 * 4 + nt]);
      o[1 * 4 + nt] = wmma_f16(pa[1], vb, o[1 * 4 + nt]);
    }

    // stage next chunk into the other buffer, then barrier
    if (havenext) {
      *(uint4*)(&kt[buf ^ 1][srow * DPAD + squad * 8]) = nxt;
      const _Float16* s8 = (const _Float16*)&nxt;
      #pragma unroll
      for (int j = 0; j < 8; ++j)
        ktT[buf ^ 1][(squad * 8 + j) * 32 + srow] = s8[j];
    }
    __syncthreads();
  }

  // write per-segment partials (combine kernel adds & normalizes)
  float* ob = opart + ((size_t)seg * TOK + qbase) * DPAD;
  #pragma unroll
  for (int qt = 0; qt < 2; ++qt)
    #pragma unroll
    for (int nt = 0; nt < 4; ++nt)
      #pragma unroll
      for (int r = 0; r < 8; ++r)
        ob[(16 * qt + r + 8 * hi) * DPAD + 16 * nt + lo] = o[qt * 4 + nt][r];
  if (lo == 0) {
    #pragma unroll
    for (int qt = 0; qt < 2; ++qt)
      #pragma unroll
      for (int r = 0; r < 8; ++r)
        lpart[(size_t)seg * TOK + qbase + 16 * qt + r + 8 * hi] = lsum[qt][r];
  }
}

// -----------------------------------------------------------------------------
// Kernel 2b: sum segment partials, normalize, emit y_hat [TOK][49] f32.
// -----------------------------------------------------------------------------
__global__ __launch_bounds__(64) void combine_norm(
    const float* __restrict__ opart, const float* __restrict__ lpart,
    float* __restrict__ yhat, int nseg) {
  int row = blockIdx.x;
  int f = threadIdx.x;
  float os = 0.f, ls = 0.f;
  for (int s = 0; s < nseg; ++s) {
    os += opart[((size_t)s * TOK + row) * DPAD + f];
    ls += lpart[(size_t)s * TOK + row];
  }
  if (f < WW) yhat[(size_t)row * WW + f] = os / ls;
}

// -----------------------------------------------------------------------------
// Kernel 3: windowed inverse mapping + pointwise 144->48 conv + bias + residual.
// y_hat_img[c][h][w] = yhat[ ((h/7)*8 + w/7)*144 + c ][ (h%7)*7 + w%7 ]
// -----------------------------------------------------------------------------
__global__ __launch_bounds__(256) void pwconv_add(
    const float* __restrict__ x, const float* __restrict__ yhat,
    const float* __restrict__ pw, const float* __restrict__ pb,
    float* __restrict__ out) {
  __shared__ float wrow[C3];
  int c = blockIdx.y;
  if (threadIdx.x < C3) wrow[threadIdx.x] = pw[c * C3 + threadIdx.x];
  __syncthreads();
  int p = blockIdx.x * 256 + threadIdx.x;
  if (p >= NPIX) return;
  int h = p / HW, w = p % HW;
  int win  = (h / 7) * 8 + (w / 7);
  int feat = (h % 7) * 7 + (w % 7);
  const float* base = yhat + (size_t)(win * C3) * WW + feat;
  float acc = pb[c];
  #pragma unroll 8
  for (int i = 0; i < C3; ++i) acc += wrow[i] * base[(size_t)i * WW];
  out[(size_t)c * NPIX + p] = x[(size_t)c * NPIX + p] + acc;
}

// -----------------------------------------------------------------------------
extern "C" void kernel_launch(void* const* d_in, const int* in_sizes, int n_in,
                              void* d_out, int out_size, void* d_ws, size_t ws_size,
                              hipStream_t stream) {
  const float* x  = (const float*)d_in[0];
  const float* w3 = (const float*)d_in[1];
  const float* w5 = (const float*)d_in[2];
  const float* w7 = (const float*)d_in[3];
  const float* pw = (const float*)d_in[4];
  const float* pb = (const float*)d_in[5];
  float* out = (float*)d_out;

  const size_t yf_b   = (size_t)TOK * DPAD * sizeof(_Float16);  // 1.18 MB
  const size_t yhat_b = (size_t)TOK * WW * sizeof(float);       // 1.81 MB
  const size_t oseg_b = (size_t)TOK * DPAD * sizeof(float);     // 2.36 MB / seg
  const size_t lseg_b = (size_t)TOK * sizeof(float);            // 36 KB / seg

  // pick the largest key-split that fits the workspace (more grid parallelism)
  int nseg = 8;
  while (nseg > 1 && yf_b + yhat_b + (size_t)nseg * (oseg_b + lseg_b) > ws_size)
    nseg >>= 1;

  char* wsp = (char*)d_ws;
  _Float16* yf   = (_Float16*)wsp;                 wsp += yf_b;
  float*    yhat = (float*)wsp;                    wsp += yhat_b;
  float*    opart = (float*)wsp;                   wsp += (size_t)nseg * oseg_b;
  float*    lpart = (float*)wsp;

  dwconv_pack<<<TOK, 64, 0, stream>>>(x, w3, w5, w7, yf);

  dim3 fg(TOK / 256, nseg);
  flash_attn<<<fg, 256, 0, stream>>>(yf, opart, lpart, TOK / nseg);

  combine_norm<<<TOK, 64, 0, stream>>>(opart, lpart, yhat, nseg);

  dim3 g((NPIX + 255) / 256, C_IN);
  pwconv_add<<<g, 256, 0, stream>>>(x, yhat, pw, pb, out);
}
